// Grouper_2903397892779
// MI455X (gfx1250) — matrix operand order repaired and checked by reading, not compiled, not run on hardware
//
#include <hip/hip_runtime.h>
#include <hip/hip_bf16.h>
#include <stddef.h>

// ---------------- CDNA5 WMMA types ----------------
typedef __attribute__((ext_vector_type(16))) __bf16 v16bf;
typedef __attribute__((ext_vector_type(8)))  float  v8f;
typedef __attribute__((ext_vector_type(4)))  unsigned int u32x4;

union BFrag {
  v16bf bf;
  u32x4 u[2];   // 32 bytes
};

#define BATCH   2
#define NPTS    16384
#define MQ      4096
#define NSAMP   32
#define CIN     64
#define KX      96      // 67 padded to 3 K-tiles of 32
#define XSTR    104     // Xt row stride in bf16 elems (96 + 8 pad, keeps 16B align)
#define Y0STR   72      // Y0 row stride in bf16 elems (64 + 8 pad)
#define BN_EPS  1e-5f

// ---------------- kernel 0: fold BN into bf16 weights ----------------
__global__ void fold_weights_kernel(
    const float* __restrict__ w0, const float* __restrict__ b0,
    const float* __restrict__ g0, const float* __restrict__ be0,
    const float* __restrict__ mu0, const float* __restrict__ va0,
    const float* __restrict__ w1, const float* __restrict__ b1,
    const float* __restrict__ g1, const float* __restrict__ be1,
    const float* __restrict__ mu1, const float* __restrict__ va1,
    __bf16* __restrict__ w0f, __bf16* __restrict__ w1f,
    float* __restrict__ bias0, float* __restrict__ bias1)
{
  const int t = blockIdx.x * blockDim.x + threadIdx.x;
  const int stride = gridDim.x * blockDim.x;
  for (int i = t; i < 64 * KX; i += stride) {
    int o = i / KX, k = i - o * KX;
    float sc = g0[o] * rsqrtf(va0[o] + BN_EPS);
    w0f[i] = (__bf16)((k < 67) ? w0[o * 67 + k] * sc : 0.0f);
  }
  for (int i = t; i < 128 * 64; i += stride) {
    int o = i >> 6;
    float sc = g1[o] * rsqrtf(va1[o] + BN_EPS);
    w1f[i] = (__bf16)(w1[i] * sc);
  }
  for (int o = t; o < 64; o += stride) {
    float sc = g0[o] * rsqrtf(va0[o] + BN_EPS);
    bias0[o] = b0[o] * sc + be0[o] - mu0[o] * sc;
  }
  for (int o = t; o < 128; o += stride) {
    float sc = g1[o] * rsqrtf(va1[o] + BN_EPS);
    bias1[o] = b1[o] * sc + be1[o] - mu1[o] * sc;
  }
}

// ---------------- kernel 1: ball query (wave32 per query point) ----------------
// Matches reference: first NSAMP indices (ascending index order) with d2 < r^2,
// tail filled with first hit (0 if the ball is empty).
__global__ __launch_bounds__(128) void ball_query_kernel(
    const float* __restrict__ xyz, const float* __restrict__ new_xyz,
    int* __restrict__ idx_out)
{
  const int lane = threadIdx.x & 31;
  const int wv   = threadIdx.x >> 5;
  const int q = blockIdx.x * 4 + wv;          // 0 .. B*M-1
  const int b = q >> 12;                      // / MQ
  const int m = q & (MQ - 1);

  const float qx = new_xyz[(b * MQ + m) * 3 + 0];
  const float qy = new_xyz[(b * MQ + m) * 3 + 1];
  const float qz = new_xyz[(b * MQ + m) * 3 + 2];
  const float r2 = 0.2f * 0.2f;

  const float* base = xyz + (size_t)b * NPTS * 3;
  int* out = idx_out + (size_t)q * NSAMP;

  int cnt = 0;
  int firstIdx = 0;
  for (int n0 = 0; n0 < NPTS; n0 += 32) {
    const int n = n0 + lane;
    const float dx = base[n * 3 + 0] - qx;
    const float dy = base[n * 3 + 1] - qy;
    const float dz = base[n * 3 + 2] - qz;
    const float d2 = dx * dx + dy * dy + dz * dz;
    const bool hit = d2 < r2;
    const unsigned bal = __builtin_amdgcn_ballot_w32(hit);
    if (bal) {
      if (cnt == 0) firstIdx = n0 + __builtin_ctz(bal);
      const int rank = __builtin_popcount(bal & ((1u << lane) - 1u));
      if (hit && (cnt + rank < NSAMP)) out[cnt + rank] = n;
      cnt += __builtin_popcount(bal);
      if (cnt >= NSAMP) break;
    }
  }
  if (cnt < NSAMP) {
    for (int s = cnt + lane; s < NSAMP; s += 32) out[s] = firstIdx;
  }
}

// ---------------- kernel 2: passthrough copy of new_xyz ----------------
__global__ void copy_kernel(const float* __restrict__ src, float* __restrict__ dst, int n) {
  int i = blockIdx.x * blockDim.x + threadIdx.x;
  if (i < n) dst[i] = src[i];
}

// ---------------- kernel 3: fused gather + MLP0 + MLP1 + maxpool ----------------
// Block: 128 threads (4 waves), tile = 2 query points = 64 positions.
// Wave w: layer0 rows [16w,16w+16), layer1 rows [32w,32w+32).
__global__ __launch_bounds__(128) void fused_group_mlp_kernel(
    const float* __restrict__ xyz, const float* __restrict__ new_xyz,
    const float* __restrict__ features, const int* __restrict__ g_idx,
    const __bf16* __restrict__ w0f, const __bf16* __restrict__ w1f,
    const float* __restrict__ bias0, const float* __restrict__ bias1,
    float* __restrict__ feats_out)
{
  __shared__ __align__(16) __bf16 Xt[64 * XSTR];   // X^T: [position][k], bf16
  __shared__ __align__(16) __bf16 Y0[64 * Y0STR];  // layer0 out: [position][c], bf16

  const int t = threadIdx.x;

  // ---- gather stage: 2 threads per position ----
  {
    const int p = t >> 1, half = t & 1;
    const int q = blockIdx.x * 2 + (p >> 5);
    const int b = q >> 12, m = q & (MQ - 1);
    const int s = p & 31;
    const int i = g_idx[(size_t)q * NSAMP + s];
    __bf16* xrow = &Xt[p * XSTR];
    if (half == 0) {
      xrow[0] = (__bf16)(xyz[((size_t)b * NPTS + i) * 3 + 0] - new_xyz[((size_t)b * MQ + m) * 3 + 0]);
      xrow[1] = (__bf16)(xyz[((size_t)b * NPTS + i) * 3 + 1] - new_xyz[((size_t)b * MQ + m) * 3 + 1]);
      xrow[2] = (__bf16)(xyz[((size_t)b * NPTS + i) * 3 + 2] - new_xyz[((size_t)b * MQ + m) * 3 + 2]);
      const float* f = features + (size_t)(b * CIN) * NPTS + i;
      #pragma unroll 8
      for (int c = 0; c < 32; ++c) xrow[3 + c] = (__bf16)f[(size_t)c * NPTS];
    } else {
      const float* f = features + (size_t)(b * CIN + 32) * NPTS + i;
      #pragma unroll 8
      for (int c = 0; c < 32; ++c) xrow[35 + c] = (__bf16)f[(size_t)c * NPTS];
      #pragma unroll
      for (int k = 67; k < KX; ++k) xrow[k] = (__bf16)0.0f;  // zero K-pad
    }
  }
  __syncthreads();

  const int lane = t & 31;
  const int wv   = t >> 5;
  const int hh   = lane >> 4;   // K-half selector for A/B fragments
  const int ln   = lane & 15;   // row (A) / column (B,D) within tile

  // ---- layer 0: 64x96 (bf16) x 96x64 -> relu -> bf16 LDS ----
  {
    BFrag A[3];
    const __bf16* wr = w0f + (16 * wv + ln) * KX;
    #pragma unroll
    for (int kt = 0; kt < 3; ++kt) {
      A[kt].u[0] = *(const u32x4*)(wr + 32 * kt + 16 * hh);
      A[kt].u[1] = *(const u32x4*)(wr + 32 * kt + 16 * hh + 8);
    }
    float bv[8];
    #pragma unroll
    for (int r = 0; r < 8; ++r) bv[r] = bias0[16 * wv + 8 * hh + r];

    #pragma unroll
    for (int nt = 0; nt < 4; ++nt) {
      v8f acc = {};
      #pragma unroll
      for (int kt = 0; kt < 3; ++kt) {
        BFrag Bf;
        const __bf16* xr = &Xt[(16 * nt + ln) * XSTR + 32 * kt + 16 * hh];
        Bf.u[0] = *(const u32x4*)(xr);
        Bf.u[1] = *(const u32x4*)(xr + 8);
        acc = __builtin_amdgcn_wmma_f32_16x16x32_bf16(
            false, A[kt].bf, false, Bf.bf, (short)0, acc, false, false);
      }
      union { __bf16 h[8]; u32x4 u; } pk;
      #pragma unroll
      for (int r = 0; r < 8; ++r)
        pk.h[r] = (__bf16)fmaxf(acc[r] + bv[r], 0.0f);
      // D[r + 8h][n] -> Y0[p][c], channels 16w+8h..+7 contiguous: one b128 store
      *(u32x4*)(&Y0[(16 * nt + ln) * Y0STR + 16 * wv + 8 * hh]) = pk.u;
    }
  }
  __syncthreads();

  // ---- layer 1: 128x64 x 64x64 + maxpool over S=32 ----
  {
    BFrag A[2][2];
    #pragma unroll
    for (int mt = 0; mt < 2; ++mt) {
      const __bf16* wr = w1f + (32 * wv + 16 * mt + ln) * 64;
      #pragma unroll
      for (int kt = 0; kt < 2; ++kt) {
        A[mt][kt].u[0] = *(const u32x4*)(wr + 32 * kt + 16 * hh);
        A[mt][kt].u[1] = *(const u32x4*)(wr + 32 * kt + 16 * hh + 8);
      }
    }
    #pragma unroll
    for (int ql = 0; ql < 2; ++ql) {       // local query within tile
      v8f acc[2][2];                        // [mt][ntp]
      #pragma unroll
      for (int ntp = 0; ntp < 2; ++ntp) {
        const int nt = 2 * ql + ntp;
        BFrag B0, B1;
        const __bf16* yr = &Y0[(16 * nt + ln) * Y0STR + 16 * hh];
        B0.u[0] = *(const u32x4*)(yr);       // K = 16h + j          (kt=0)
        B0.u[1] = *(const u32x4*)(yr + 8);
        B1.u[0] = *(const u32x4*)(yr + 32);  // K = 32 + 16h + j     (kt=1)
        B1.u[1] = *(const u32x4*)(yr + 40);
        #pragma unroll
        for (int mt = 0; mt < 2; ++mt) {
          v8f a = {};
          a = __builtin_amdgcn_wmma_f32_16x16x32_bf16(
              false, A[mt][0].bf, false, B0.bf, (short)0, a, false, false);
          a = __builtin_amdgcn_wmma_f32_16x16x32_bf16(
              false, A[mt][1].bf, false, B1.bf, (short)0, a, false, false);
          acc[mt][ntp] = a;
        }
      }
      // maxpool: max over the two 16-position tiles + cross-lane max over n.
      // relu(max(y)+bias) == max over relu(y+bias) since bias is per-channel
      // and relu is monotone.
      const int qg = blockIdx.x * 2 + ql;
      const int b = qg >> 12, m = qg & (MQ - 1);
      #pragma unroll
      for (int mt = 0; mt < 2; ++mt) {
        float vmx[8];
        #pragma unroll
        for (int r = 0; r < 8; ++r) {
          float v = fmaxf(acc[mt][0][r], acc[mt][1][r]);
          v = fmaxf(v, __shfl_xor(v, 1, 32));
          v = fmaxf(v, __shfl_xor(v, 2, 32));
          v = fmaxf(v, __shfl_xor(v, 4, 32));
          v = fmaxf(v, __shfl_xor(v, 8, 32));
          vmx[r] = v;
        }
        if (ln == 0) {   // lanes 0 (h=0) and 16 (h=1) write 8 channels each
          const int c0 = 32 * wv + 16 * mt + 8 * hh;
          float* op = feats_out + ((size_t)b * 128 + c0) * MQ + m;
          #pragma unroll
          for (int r = 0; r < 8; ++r)
            op[(size_t)r * MQ] = fmaxf(vmx[r] + bias1[c0 + r], 0.0f);
        }
      }
    }
  }
}

// ---------------- launcher ----------------
extern "C" void kernel_launch(void* const* d_in, const int* in_sizes, int n_in,
                              void* d_out, int out_size, void* d_ws, size_t ws_size,
                              hipStream_t stream) {
  const float* xyz      = (const float*)d_in[0];
  const float* new_xyz  = (const float*)d_in[1];
  const float* features = (const float*)d_in[2];
  const float* w0  = (const float*)d_in[3];
  const float* b0  = (const float*)d_in[4];
  const float* g0  = (const float*)d_in[5];
  const float* be0 = (const float*)d_in[6];
  const float* mu0 = (const float*)d_in[7];
  const float* va0 = (const float*)d_in[8];
  const float* w1  = (const float*)d_in[9];
  const float* b1  = (const float*)d_in[10];
  const float* g1  = (const float*)d_in[11];
  const float* be1 = (const float*)d_in[12];
  const float* mu1 = (const float*)d_in[13];
  const float* va1 = (const float*)d_in[14];

  // workspace layout (all 256B-aligned):
  //   [0, 1 MB)              : ball-query indices (B*M*32 int32)
  //   [+0,   12288)          : W0f bf16 64x96 (BN-folded, K-padded)
  //   [+,    16384)          : W1f bf16 128x64 (BN-folded)
  //   [+,    256) / (+, 512) : bias0 f32[64], bias1 f32[128]
  char* ws = (char*)d_ws;
  int*    idx   = (int*)ws;
  __bf16* w0f   = (__bf16*)(ws + 1048576);
  __bf16* w1f   = (__bf16*)(ws + 1048576 + 12288);
  float*  bias0 = (float*)(ws + 1048576 + 12288 + 16384);
  float*  bias1 = (float*)(ws + 1048576 + 12288 + 16384 + 256);

  float* out_newxyz = (float*)d_out;                 // B*M*3 = 24576 floats
  float* out_feats  = (float*)d_out + BATCH * MQ * 3; // B*128*M floats

  fold_weights_kernel<<<56, 256, 0, stream>>>(
      w0, b0, g0, be0, mu0, va0, w1, b1, g1, be1, mu1, va1,
      w0f, w1f, bias0, bias1);

  ball_query_kernel<<<(BATCH * MQ) / 4, 128, 0, stream>>>(xyz, new_xyz, idx);

  copy_kernel<<<(BATCH * MQ * 3 + 255) / 256, 256, 0, stream>>>(
      new_xyz, out_newxyz, BATCH * MQ * 3);

  fused_group_mlp_kernel<<<(BATCH * MQ) / 2, 128, 0, stream>>>(
      xyz, new_xyz, features, idx, w0f, w1f, bias0, bias1, out_feats);
}
